// SystemDEQ_44006234914917
// MI455X (gfx1250) — compile-verified, matrix operand order checked by reading, not run on hardware
//
#include <hip/hip_runtime.h>
#include <hip/hip_bf16.h>
#include <math.h>

// ---------------------------------------------------------------------------
// Types for CDNA5 WMMA (gfx1250, wave32)
// ---------------------------------------------------------------------------
typedef __attribute__((ext_vector_type(16))) __bf16 v16bf;
typedef __attribute__((ext_vector_type(8)))  __bf16 v8bf;
typedef __attribute__((ext_vector_type(8)))  float  v8f;

#define B_    512   // batch
#define D_    512   // hidden
#define IN_   64    // input dim
#define DIN_  576   // hidden + input
#define KMAX_ 100   // broyden max_iter
#define OUTD_ 32    // output dim
#define TOL_  1e-6f

__device__ __forceinline__ __bf16 f32_to_bf16(float f) {
    unsigned int u = __builtin_bit_cast(unsigned int, f);
    unsigned int r = u + 0x7FFFu + ((u >> 16) & 1u);   // round-to-nearest-even
    unsigned short h = (unsigned short)(r >> 16);
    return __builtin_bit_cast(__bf16, h);
}

__device__ __forceinline__ v16bf join16(v8bf lo, v8bf hi) {
    v16bf r;
#pragma unroll
    for (int i = 0; i < 8; ++i) { r[i] = lo[i]; r[i + 8] = hi[i]; }
    return r;
}

// Load one 16x32 bf16 fragment (ISA 7.12.2 layout) from row-major src
// (leading dim K): per lane two contiguous 16B chunks at +0 and +16 elems.
__device__ __forceinline__ v16bf load_frag(const __bf16* p) {
    return join16(*(const v8bf*)(p), *(const v8bf*)(p + 16));
}

// ---------------------------------------------------------------------------
// WMMA GEMM, register-blocked + double-buffered.
// Each wave computes a 16 x (16*NT) strip of out[M,N] =
//   epilogue(A[M,K] @ WT[N,K]^T + bias[N]).
// A and WT are bf16 row-major with leading dim K (WT pre-transposed so both
// fragments are contiguous 16B loads). K is compile-time: the NT B-fragments
// and all K-steps address off ONE base pointer with immediate offsets
// (global_load_b128 clauses), and the K-loop fully unrolls so the
// double-buffer fragment arrays live in registers.
// Pipeline: loads for step s+1 are issued BEFORE the WMMAs of step s, so each
// WMMA waits only for iteration-old data (partial s_wait_loadcnt, ~10 loads
// in flight) instead of draining the load queue.
// Fragment layout per CDNA5 ISA 7.12.2 (16-bit A 16x32):
//   lane L: row/col = L&15, half = L>>4
//   v16 elements 0..7  <- K = kb + half*8 + (0..7)
//   v16 elements 8..15 <- K = kb + 16 + half*8 + (0..7)
// C/D: v8f element i -> row = tile_m*16 + i + 8*half, col = tile_n*16 + (L&15)
// MODE 0: tanh -> bf16 out; MODE 1: linear -> f32 out; MODE 2: linear -> bf16
// ---------------------------------------------------------------------------
template <int MODE, int NT, int K>
__global__ __launch_bounds__(256) void wmma_gemm(
    const __bf16* __restrict__ A, const __bf16* __restrict__ WT,
    const float* __restrict__ bias, void* __restrict__ out,
    int M, int N)
{
    const int lane = threadIdx.x & 31;
    const int wave = (blockIdx.x * blockDim.x + threadIdx.x) >> 5;
    const int wn   = (N >> 4) / NT;              // wave-strips across N
    const int strips = (M >> 4) * wn;
    if (wave >= strips) return;                  // uniform per wave: EXEC all-1s
    const int tm  = wave / wn;
    const int tn0 = (wave % wn) * NT;
    const int half = lane >> 4;
    const int l15  = lane & 15;

    const __bf16* ap = A  + (size_t)(tm * 16 + l15) * K + half * 8;
    const __bf16* bp = WT + (size_t)(tn0 * 16 + l15) * K + half * 8;

    v8f acc[NT];
#pragma unroll
    for (int t = 0; t < NT; ++t) acc[t] = (v8f){};

    constexpr int NSTEP = K / 32;
    v16bf av[2];
    v16bf bv[2][NT];

    // prologue: stage step 0
    av[0] = load_frag(ap);
#pragma unroll
    for (int t = 0; t < NT; ++t) bv[0][t] = load_frag(bp + t * 16 * K);

#pragma unroll
    for (int s = 0; s < NSTEP; ++s) {
        const int cur = s & 1, nxt = cur ^ 1;
        if (s + 1 < NSTEP) {                      // prefetch step s+1
            const int kb = (s + 1) * 32;
            av[nxt] = load_frag(ap + kb);
#pragma unroll
            for (int t = 0; t < NT; ++t)
                bv[nxt][t] = load_frag(bp + t * 16 * K + kb);
        }
#pragma unroll
        for (int t = 0; t < NT; ++t)
            acc[t] = __builtin_amdgcn_wmma_f32_16x16x32_bf16(
                false, av[cur], false, bv[cur][t], (short)0, acc[t], false, false);
    }

#pragma unroll
    for (int t = 0; t < NT; ++t) {
        const int n = (tn0 + t) * 16 + l15;
        const float bn = bias ? bias[n] : 0.0f;
#pragma unroll
        for (int i = 0; i < 8; ++i) {
            const int m = tm * 16 + i + half * 8;
            float v = acc[t][i] + bn;
            if (MODE == 0) {
                ((__bf16*)out)[(size_t)m * N + n] = f32_to_bf16(tanhf(v));
            } else if (MODE == 1) {
                ((float*)out)[(size_t)m * N + n] = v;
            } else {
                ((__bf16*)out)[(size_t)m * N + n] = f32_to_bf16(v);
            }
        }
    }
}

// ---------------------------------------------------------------------------
// Prep kernels
// ---------------------------------------------------------------------------
__global__ void k_transpose_bf16(const float* __restrict__ W,
                                 __bf16* __restrict__ WT, int K, int N) {
    int idx = blockIdx.x * blockDim.x + threadIdx.x;     // over N*K
    if (idx >= N * K) return;
    int n = idx / K, kk = idx % K;
    WT[(size_t)n * K + kk] = f32_to_bf16(W[(size_t)kk * N + n]);
}

__global__ void k_fill_u(const float* __restrict__ u, __bf16* __restrict__ A0) {
    int idx = blockIdx.x * blockDim.x + threadIdx.x;     // over B*IN
    if (idx >= B_ * IN_) return;
    int b = idx / IN_, j = idx % IN_;
    A0[(size_t)b * DIN_ + D_ + j] = f32_to_bf16(u[(size_t)b * IN_ + j]);
}

__global__ void k_zero_f32(float* __restrict__ x, int n) {
    int idx = blockIdx.x * blockDim.x + threadIdx.x;
    if (idx < n) x[idx] = 0.0f;
}

__global__ void k_a0z(const float* __restrict__ x, __bf16* __restrict__ A0) {
    int idx = blockIdx.x * blockDim.x + threadIdx.x;     // over B*D
    if (idx >= B_ * D_) return;
    int b = idx / D_, d = idx % D_;
    A0[(size_t)b * DIN_ + d] = f32_to_bf16(x[idx]);
}

// gx = fbuf - x   (initial residual; x==0 at start)
__global__ void k_init_gx(const float* __restrict__ fbuf,
                          const float* __restrict__ x, float* __restrict__ gx) {
    int idx = blockIdx.x * blockDim.x + threadIdx.x;
    if (idx < B_ * D_) gx[idx] = fbuf[idx] - x[idx];
}

// ---------------------------------------------------------------------------
// Broyden kernels
// ---------------------------------------------------------------------------
// conv[b] = ||gx[b,:]|| < tol
__global__ void k_norm(const float* __restrict__ gx, float* __restrict__ conv) {
    __shared__ float red[256];
    const int b = blockIdx.x;
    float s = 0.0f;
    for (int d = threadIdx.x; d < D_; d += blockDim.x) {
        float v = gx[(size_t)b * D_ + d];
        s += v * v;
    }
    red[threadIdx.x] = s; __syncthreads();
    for (int off = 128; off > 0; off >>= 1) {
        if ((int)threadIdx.x < off) red[threadIdx.x] += red[threadIdx.x + off];
        __syncthreads();
    }
    if (threadIdx.x == 0) conv[b] = (sqrtf(red[0]) < TOL_) ? 1.0f : 0.0f;
}

// c[kk,b] = dot(mat[kk,b,:], vec[b,:])  for kk in [0, gridDim.x)
__global__ void k_dots(const float* __restrict__ mat,
                       const float* __restrict__ vec, float* __restrict__ c) {
    __shared__ float red[256];
    const int kk = blockIdx.x, b = blockIdx.y;
    const float* row = mat + ((size_t)kk * B_ + b) * D_;
    const float* v   = vec + (size_t)b * D_;
    float s = 0.0f;
    for (int d = threadIdx.x; d < D_; d += blockDim.x) s += row[d] * v[d];
    red[threadIdx.x] = s; __syncthreads();
    for (int off = 128; off > 0; off >>= 1) {
        if ((int)threadIdx.x < off) red[threadIdx.x] += red[threadIdx.x + off];
        __syncthreads();
    }
    if (threadIdx.x == 0) c[(size_t)kk * B_ + b] = red[0];
}

// update = conv ? 0 : (gx - sum_k Us[k]*c1[k,b]);  dx=update; xnew=x+update;
// A0 z-part = bf16(xnew)
__global__ void k_update(const float* __restrict__ gx, const float* __restrict__ x,
                         const float* __restrict__ Us, const float* __restrict__ c,
                         const float* __restrict__ conv,
                         float* __restrict__ dx, float* __restrict__ xnew,
                         __bf16* __restrict__ A0, int kcur) {
    int idx = blockIdx.x * blockDim.x + threadIdx.x;
    if (idx >= B_ * D_) return;
    int b = idx / D_, d = idx % D_;
    float s = 0.0f;
    for (int kk = 0; kk < kcur; ++kk)
        s += Us[((size_t)kk * B_ + b) * D_ + d] * c[(size_t)kk * B_ + b];
    float upd = (conv[b] != 0.0f) ? 0.0f : (gx[idx] - s);
    dx[idx] = upd;
    float xn = x[idx] + upd;
    xnew[idx] = xn;
    A0[(size_t)b * DIN_ + d] = f32_to_bf16(xn);
}

// gx_new = fbuf - xnew (stored back into fbuf); dgx = gx_new - gx
__global__ void k_postf(float* __restrict__ fbuf, const float* __restrict__ xnew,
                        const float* __restrict__ gx, float* __restrict__ dgx) {
    int idx = blockIdx.x * blockDim.x + threadIdx.x;
    if (idx >= B_ * D_) return;
    float t = fbuf[idx] - xnew[idx];
    fbuf[idx] = t;
    dgx[idx] = t - gx[idx];
}

// vT = -dx + sum_k VTs[k]*c2[k,b]
__global__ void k_vt(const float* __restrict__ dx, const float* __restrict__ VTs,
                     const float* __restrict__ c, float* __restrict__ vT, int kcur) {
    int idx = blockIdx.x * blockDim.x + threadIdx.x;
    if (idx >= B_ * D_) return;
    int b = idx / D_, d = idx % D_;
    float s = 0.0f;
    for (int kk = 0; kk < kcur; ++kk)
        s += VTs[((size_t)kk * B_ + b) * D_ + d] * c[(size_t)kk * B_ + b];
    vT[idx] = -dx[idx] + s;
}

// denom[b] = sum_d vT[b,d]*dgx[b,d]
__global__ void k_denom(const float* __restrict__ vT, const float* __restrict__ dgx,
                        float* __restrict__ denom) {
    __shared__ float red[256];
    const int b = blockIdx.x;
    float s = 0.0f;
    for (int d = threadIdx.x; d < D_; d += blockDim.x)
        s += vT[(size_t)b * D_ + d] * dgx[(size_t)b * D_ + d];
    red[threadIdx.x] = s; __syncthreads();
    for (int off = 128; off > 0; off >>= 1) {
        if ((int)threadIdx.x < off) red[threadIdx.x] += red[threadIdx.x + off];
        __syncthreads();
    }
    if (threadIdx.x == 0) denom[b] = red[0];
}

// u = safe&&!conv ? (dx + dgx - sum_k Us[k]*c3[k,b]) / denom : 0
// Us[kcur]=u; VTs[kcur]= conv?0:vT; x = conv?x:xnew; gx = conv?gx:gx_new(fbuf)
__global__ void k_final(const float* __restrict__ dx, const float* __restrict__ dgx,
                        const float* __restrict__ vT, const float* __restrict__ c,
                        const float* __restrict__ denom, const float* __restrict__ conv,
                        const float* __restrict__ xnew, const float* __restrict__ gxn,
                        float* __restrict__ Us, float* __restrict__ VTs,
                        float* __restrict__ x, float* __restrict__ gx, int kcur) {
    int idx = blockIdx.x * blockDim.x + threadIdx.x;
    if (idx >= B_ * D_) return;
    int b = idx / D_, d = idx % D_;
    float s = 0.0f;
    for (int kk = 0; kk < kcur; ++kk)
        s += Us[((size_t)kk * B_ + b) * D_ + d] * c[(size_t)kk * B_ + b];
    float unum = dx[idx] + dgx[idx] - s;
    float den  = denom[b];
    bool  safe = fabsf(den) > 1e-12f;
    bool  cv   = (conv[b] != 0.0f);
    float uval = (safe && !cv) ? (unum / den) : 0.0f;
    Us [((size_t)kcur * B_ + b) * D_ + d] = uval;
    VTs[((size_t)kcur * B_ + b) * D_ + d] = cv ? 0.0f : vT[idx];
    if (!cv) { x[idx] = xnew[idx]; gx[idx] = gxn[idx]; }
}

// ---------------------------------------------------------------------------
// Host orchestration: whole Broyden solve enqueued on one stream (graph-safe)
// ---------------------------------------------------------------------------
extern "C" void kernel_launch(void* const* d_in, const int* in_sizes, int n_in,
                              void* d_out, int out_size, void* d_ws, size_t ws_size,
                              hipStream_t stream) {
    const float* u    = (const float*)d_in[0];   // [512,64]
    const float* W0   = (const float*)d_in[1];   // [576,512]
    const float* b0   = (const float*)d_in[2];   // [512]
    const float* Ws   = (const float*)d_in[3];   // [7,512,512]
    const float* bs   = (const float*)d_in[4];   // [7,512]
    const float* Wout = (const float*)d_in[5];   // [512,32]
    const float* bout = (const float*)d_in[6];   // [32]
    float* outp = (float*)d_out;                 // [512,32] f32

    // Workspace layout (256B aligned slabs)
    char* p = (char*)d_ws;
    auto alloc = [&](size_t bytes) -> void* {
        void* r = (void*)p;
        p += (bytes + 255) & ~(size_t)255;
        return r;
    };
    const size_t BD = (size_t)B_ * D_;
    float*  x    = (float*) alloc(BD * 4);
    float*  gx   = (float*) alloc(BD * 4);
    float*  xnew = (float*) alloc(BD * 4);
    float*  dx   = (float*) alloc(BD * 4);
    float*  fbuf = (float*) alloc(BD * 4);     // f output -> then gx_new
    float*  dgx  = (float*) alloc(BD * 4);
    float*  vT   = (float*) alloc(BD * 4);
    float*  conv = (float*) alloc((size_t)B_ * 4);
    float*  denom= (float*) alloc((size_t)B_ * 4);
    float*  cbuf = (float*) alloc((size_t)KMAX_ * B_ * 4);
    __bf16* A0   = (__bf16*)alloc((size_t)B_ * DIN_ * 2);
    __bf16* hA   = (__bf16*)alloc(BD * 2);
    __bf16* hB   = (__bf16*)alloc(BD * 2);
    __bf16* WT0  = (__bf16*)alloc((size_t)D_ * DIN_ * 2);
    __bf16* WTs  = (__bf16*)alloc((size_t)7 * D_ * D_ * 2);
    __bf16* WTo  = (__bf16*)alloc((size_t)OUTD_ * D_ * 2);
    float*  Us   = (float*) alloc((size_t)KMAX_ * BD * 4);   // ~100 MB
    float*  VTs  = (float*) alloc((size_t)KMAX_ * BD * 4);   // ~100 MB

    const int EW = (B_ * D_ + 255) / 256;        // elementwise grid (1024)

    // GEMM grids: waves = (M/16) * (N/16/NT); 8 waves per 256-thread block
    const int GRID_H = (32 * (32 / 4)) / 8;      // 512x512, NT=4 -> 32 blocks
    const int GRID_O = (32 * (2 / 2)) / 8;       // 512x32,  NT=2 -> 4 blocks

    // ---- prep: bf16 transposed weights, constant u-part of A0, x = 0 ----
    k_transpose_bf16<<<(D_ * DIN_ + 255) / 256, 256, 0, stream>>>(W0, WT0, DIN_, D_);
    for (int i = 0; i < 7; ++i)
        k_transpose_bf16<<<(D_ * D_ + 255) / 256, 256, 0, stream>>>(
            Ws + (size_t)i * D_ * D_, WTs + (size_t)i * D_ * D_, D_, D_);
    k_transpose_bf16<<<(OUTD_ * D_ + 255) / 256, 256, 0, stream>>>(Wout, WTo, D_, OUTD_);
    k_fill_u<<<(B_ * IN_ + 255) / 256, 256, 0, stream>>>(u, A0);
    k_zero_f32<<<EW, 256, 0, stream>>>(x, B_ * D_);

    // MLP: A0 -> fbuf(f32) or last layer to bf16 buffer
    auto mlp = [&](int lastMode, void* lastOut) {
        wmma_gemm<0, 4, DIN_><<<GRID_H, 256, 0, stream>>>(A0, WT0, b0, hA, 512, 512);
        __bf16* cur = hA; __bf16* nxt = hB;
        for (int i = 0; i < 6; ++i) {
            wmma_gemm<0, 4, D_><<<GRID_H, 256, 0, stream>>>(
                cur, WTs + (size_t)i * D_ * D_, bs + (size_t)i * D_, nxt, 512, 512);
            __bf16* t = cur; cur = nxt; nxt = t;
        }
        if (lastMode == 1)
            wmma_gemm<1, 4, D_><<<GRID_H, 256, 0, stream>>>(
                cur, WTs + (size_t)6 * D_ * D_, bs + (size_t)6 * D_, lastOut, 512, 512);
        else
            wmma_gemm<2, 4, D_><<<GRID_H, 256, 0, stream>>>(
                cur, WTs + (size_t)6 * D_ * D_, bs + (size_t)6 * D_, lastOut, 512, 512);
    };

    // ---- gx0 = f(0) - 0 ----
    k_a0z<<<EW, 256, 0, stream>>>(x, A0);
    mlp(1, fbuf);
    k_init_gx<<<EW, 256, 0, stream>>>(fbuf, x, gx);

    // ---- Broyden iterations ----
    for (int k = 0; k < KMAX_; ++k) {
        k_norm<<<B_, 256, 0, stream>>>(gx, conv);
        if (k > 0) k_dots<<<dim3(k, B_), 256, 0, stream>>>(VTs, gx, cbuf);   // c1
        k_update<<<EW, 256, 0, stream>>>(gx, x, Us, cbuf, conv, dx, xnew, A0, k);
        mlp(1, fbuf);                                                        // f(x_new)
        k_postf<<<EW, 256, 0, stream>>>(fbuf, xnew, gx, dgx);
        if (k > 0) k_dots<<<dim3(k, B_), 256, 0, stream>>>(Us, dx, cbuf);    // c2
        k_vt<<<EW, 256, 0, stream>>>(dx, VTs, cbuf, vT, k);
        k_denom<<<B_, 256, 0, stream>>>(vT, dgx, denom);
        if (k > 0) k_dots<<<dim3(k, B_), 256, 0, stream>>>(VTs, dgx, cbuf);  // c3
        k_final<<<EW, 256, 0, stream>>>(dx, dgx, vT, cbuf, denom, conv,
                                        xnew, fbuf, Us, VTs, x, gx, k);
    }

    // ---- z_star = f(x); out = z_star @ Wout + bout ----
    k_a0z<<<EW, 256, 0, stream>>>(x, A0);
    mlp(2, hB);                                     // z_star as bf16 in hB
    wmma_gemm<1, 2, D_><<<GRID_O, 256, 0, stream>>>(hB, WTo, bout, outp, 512, 32);
}
